// SA1_012_9861244912095
// MI455X (gfx1250) — matrix operand order repaired and checked by reading, not compiled, not run on hardware
//
#include <hip/hip_runtime.h>

// ---------------------------------------------------------------------------
// MI455X / gfx1250 implementation.
// Dominant cost: 2 attention blocks, 64 heads of (1024 x 1024 x 32) QK^T + PV
// => ~17 GFLOP. Implemented as register-resident flash attention using
// v_wmma_f32_16x16x32_f16 (f16 operands, f32 accumulate), one wave32 per
// 16-query tile, transposed score layout so softmax stats are per-lane.
//   - Q is pre-scaled by log2(e)/32 so scores are base-2 logits: exp -> raw
//     v_exp_f32 (TRANS, co-executes with XDL WMMA), no per-score multiplies.
//   - P^T fragment packing uses v_cvt_pkrtz + one dword shuffle per row pair
//     (8 ds ops / 32-key block instead of 16).
//   - row-sum halves are combined once after the key loop (linear in alpha).
// Pointwise grouped-linear MLPs run as unrolled f32 FMAs with LDS weights.
// ---------------------------------------------------------------------------

#define NBATCH 32
#define NTOK   1024
#define DHEAD  32
#define TOK    (NBATCH*NTOK)

// fold softmax's double scale (1/D) and log2(e) into the stored Q values
#define QSCALE (1.4426950408889634f / 32.0f)

typedef __attribute__((ext_vector_type(16))) _Float16 v16h;
typedef __attribute__((ext_vector_type(8)))  float    v8f;

__device__ __forceinline__ float lrelu(float x){ return x > 0.0f ? x : 0.01f*x; }

__device__ __forceinline__ void lds_copy(float* dst, const float* __restrict__ src, int n){
  for (int i = threadIdx.x; i < n; i += blockDim.x) dst[i] = src[i];
}

// grouped linear: y[g*Co+o] = b[g*Co+o] + sum_i W[(g*Co+o)*Ci+i] * x[g*Ci+i]
template<int G, int Co, int Ci, bool ACT>
__device__ __forceinline__ void glinear(const float* __restrict__ x, float* __restrict__ y,
                                        const float* W, const float* b){
#pragma unroll
  for (int g = 0; g < G; ++g) {
#pragma unroll
    for (int o = 0; o < Co; ++o) {
      float a = b[g*Co + o];
      const float* w = W + (g*Co + o)*Ci;
#pragma unroll
      for (int i = 0; i < Ci; ++i) a = fmaf(w[i], x[g*Ci + i], a);
      y[g*Co + o] = ACT ? lrelu(a) : a;
    }
  }
}

// qkv glinear (6 groups over tiled(x_e,3)) + scatter q/k (f16), v (f32), v^T (f16)
// q is stored pre-scaled by QSCALE (see header comment).
__device__ __forceinline__ void qkv_project_store(
    const float* __restrict__ xe, const float* __restrict__ qW, const float* __restrict__ qb,
    int b, int n,
    _Float16* __restrict__ qh, _Float16* __restrict__ kh,
    _Float16* __restrict__ vt, float* __restrict__ vf)
{
  float out[192];
#pragma unroll
  for (int g = 0; g < 6; ++g) {
    const float* xg = xe + (g & 1)*32;          // tiled input: group g reads xe[(g%2)*32..]
#pragma unroll
    for (int o = 0; o < 32; ++o) {
      float a = qb[g*32 + o];
      const float* w = qW + (g*32 + o)*32;
#pragma unroll
      for (int i = 0; i < 32; ++i) a = fmaf(w[i], xg[i], a);
      out[g*32 + o] = a;
    }
  }
#pragma unroll
  for (int hd = 0; hd < 2; ++hd) {
    int hg   = b*2 + hd;
    int base = (hg*NTOK + n)*DHEAD;
#pragma unroll
    for (int d0 = 0; d0 < DHEAD; ++d0) {
      qh[base + d0] = (_Float16)(out[hd*32 + d0] * QSCALE);   // s=0 (pre-scaled)
      kh[base + d0] = (_Float16)out[64 + hd*32 + d0];         // s=1
      float vv = out[128 + hd*32 + d0];                       // s=2
      vf[base + d0] = vv;
      vt[(hg*DHEAD + d0)*NTOK + n] = (_Float16)vv;            // transposed copy for PV A-frags
    }
  }
}

// ---------------------------------------------------------------------------
// K1: x0 -> invert last chan -> embedding chain -> x_e -> sdp1 qkv
// ---------------------------------------------------------------------------
__global__ void __launch_bounds__(256)
k_emb_qkv(const float* __restrict__ x0,
          const float* __restrict__ e1W, const float* __restrict__ e1b,
          const float* __restrict__ e2W, const float* __restrict__ e2b,
          const float* __restrict__ e3W, const float* __restrict__ e3b,
          const float* __restrict__ e4W, const float* __restrict__ e4b,
          const float* __restrict__ e5W, const float* __restrict__ e5b,
          const float* __restrict__ qW,  const float* __restrict__ qb,
          _Float16* __restrict__ qh, _Float16* __restrict__ kh,
          _Float16* __restrict__ vt, float* __restrict__ vf)
{
  __shared__ float s[8704];
  float* sE1W = s;        float* sE1b = s + 192;
  float* sE2W = s + 256;  float* sE2b = s + 2304;
  float* sE3W = s + 2368; float* sE3b = s + 4416;
  float* sE4W = s + 4480; float* sE4b = s + 6528;
  float* sE5W = s + 6592; float* sE5b = s + 8640;
  lds_copy(sE1W, e1W, 192);  lds_copy(sE1b, e1b, 64);
  lds_copy(sE2W, e2W, 2048); lds_copy(sE2b, e2b, 64);
  lds_copy(sE3W, e3W, 2048); lds_copy(sE3b, e3b, 64);
  lds_copy(sE4W, e4W, 2048); lds_copy(sE4b, e4b, 64);
  lds_copy(sE5W, e5W, 2048); lds_copy(sE5b, e5b, 64);
  __syncthreads();

  int t = blockIdx.x*256 + threadIdx.x;
  int b = t >> 10, n = t & 1023;
  float xin[3];
  xin[0] = x0[t*3 + 0];
  xin[1] = x0[t*3 + 1];
  xin[2] = 1.0f / x0[t*3 + 2];

  float h[64], h2[64];
  // e1: (2,32,3) over tiled(x0,2): both groups read the same 3 inputs
#pragma unroll
  for (int g = 0; g < 2; ++g)
#pragma unroll
    for (int o = 0; o < 32; ++o) {
      const float* w = sE1W + (g*32 + o)*3;
      float a = sE1b[g*32 + o];
      a = fmaf(w[0], xin[0], a);
      a = fmaf(w[1], xin[1], a);
      a = fmaf(w[2], xin[2], a);
      h[g*32 + o] = lrelu(a);
    }
  glinear<2,32,32,true >(h,  h2, sE2W, sE2b);
  glinear<2,32,32,true >(h2, h,  sE3W, sE3b);
  glinear<2,32,32,true >(h,  h2, sE4W, sE4b);
  glinear<2,32,32,false>(h2, h,  sE5W, sE5b);   // x_e
  qkv_project_store(h, qW, qb, b, n, qh, kh, vt, vf);
}

// ---------------------------------------------------------------------------
// K2: flash attention, one wave per 16-query tile, 64 heads total.
// S^T = K_tile(A) x Q^T(B)  -> scores for one query live in one lane (+lane^16)
// O^T = V^T(A)  x P^T(B)    -> accumulated in two 16x16 f32 fragments
// ---------------------------------------------------------------------------
__global__ void __launch_bounds__(128)
k_attn(const _Float16* __restrict__ qh, const _Float16* __restrict__ kh,
       const _Float16* __restrict__ vt, float* __restrict__ attn)
{
  const int lane = threadIdx.x & 31;
  const int wv   = threadIdx.x >> 5;
  const int tile = blockIdx.x*4 + wv;          // 0..4095
  const int hg   = tile >> 6;                  // head-group 0..63
  const int q0   = (tile & 63) << 4;           // query tile base
  const int hp   = (lane < 16) ? 0 : 1;        // half-wave id
  const int l16  = lane & 15;

  // Q^T B-fragment, resident for the whole key loop.
  // B layout: lanes 0-15 hold K(dim)=0..15 (2 halves/VGPR), lanes 16-31 dims 16..31.
  union { v16h v; _Float16 h[16]; float4 f4[2]; } qf;
  {
    const _Float16* p = qh + (hg*NTOK + (q0 + l16))*DHEAD + hp*16;
    qf.f4[0] = *(const float4*)(p);
    qf.f4[1] = *(const float4*)(p + 8);
  }

  v8f o_lo = {}; v8f o_hi = {};
  float m = -3.0e38f, lsum = 0.0f;             // base-2 domain stats

  for (int kb = 0; kb < NTOK/32; ++kb) {
    // --- S^T for two 16-key tiles (keys kb*32 .. kb*32+31), base-2 logits ---
    v8f s0, s1;
#pragma unroll
    for (int tt = 0; tt < 2; ++tt) {
      // K_tile A-frag: lane row = key; lanes<16 get dims 0-15, lanes>=16 dims 16-31
      union { v16h v; _Float16 h[16]; float4 f4[2]; } kf;
      int key = kb*32 + tt*16 + l16;
      int dst = hp*16;
      const _Float16* p = kh + (hg*NTOK + key)*DHEAD;
      kf.f4[0] = *(const float4*)(p + dst);
      kf.f4[1] = *(const float4*)(p + dst + 8);
      v8f z = {};
      v8f r = __builtin_amdgcn_wmma_f32_16x16x32_f16(false, kf.v, false, qf.v,
                                                     (short)0, z, false, false);
      if (tt == 0) s0 = r; else s1 = r;
    }

    // --- online softmax (per-lane stats, query = lane%16, one shuffle for max) ---
    float mx = -3.0e38f;
#pragma unroll
    for (int r = 0; r < 8; ++r) mx = fmaxf(mx, fmaxf(s0[r], s1[r]));
    mx = fmaxf(mx, __shfl_xor(mx, 16, 32));
    float mnew  = fmaxf(m, mx);
    float alpha = __builtin_amdgcn_exp2f(m - mnew);

    float p0[8], p1[8];
    float rs = 0.0f;                            // per-half partial row sum
#pragma unroll
    for (int r = 0; r < 8; ++r) {
      p0[r] = __builtin_amdgcn_exp2f(s0[r] - mnew);
      p1[r] = __builtin_amdgcn_exp2f(s1[r] - mnew);
      rs += p0[r] + p1[r];
    }
    lsum = lsum*alpha + rs;                     // halves combined after loop
    m = mnew;

    // --- pack P^T (32 keys x 16 queries) into B-fragment ---
    // pack (p0,p1) -> one f16x2 dword, one xor-16 shuffle covers both halves:
    // lanes<16 need keys 0-15 (frag0 own rows + partner rows)
    // lanes>=16 need keys 16-31 (frag1 partner rows + own rows)
    union { v16h v; unsigned short us[16]; } pt;
#pragma unroll
    for (int r = 0; r < 8; ++r) {
      int pk = __builtin_bit_cast(int, __builtin_amdgcn_cvt_pkrtz(p0[r], p1[r]));
      int sh = __shfl_xor(pk, 16, 32);
      unsigned short own_lo = (unsigned short)( pk        & 0xffff);
      unsigned short own_hi = (unsigned short)(((unsigned)pk) >> 16);
      unsigned short sh_lo  = (unsigned short)( sh        & 0xffff);
      unsigned short sh_hi  = (unsigned short)(((unsigned)sh) >> 16);
      pt.us[r]     = hp ? sh_hi  : own_lo;
      pt.us[r + 8] = hp ? own_hi : sh_lo;
    }

    // --- rescale O and accumulate O^T += V^T x P^T (two dim-halves) ---
#pragma unroll
    for (int r = 0; r < 8; ++r) { o_lo[r] *= alpha; o_hi[r] *= alpha; }
#pragma unroll
    for (int dh = 0; dh < 2; ++dh) {
      union { v16h v; _Float16 h[16]; float4 f4[2]; } vfr;
      int dim = dh*16 + l16;
      int kst = hp*16;
      const _Float16* p = vt + (hg*DHEAD + dim)*NTOK + kb*32;
      vfr.f4[0] = *(const float4*)(p + kst);
      vfr.f4[1] = *(const float4*)(p + kst + 8);
      if (dh == 0)
        o_lo = __builtin_amdgcn_wmma_f32_16x16x32_f16(false, vfr.v, false, pt.v,
                                                      (short)0, o_lo, false, false);
      else
        o_hi = __builtin_amdgcn_wmma_f32_16x16x32_f16(false, vfr.v, false, pt.v,
                                                      (short)0, o_hi, false, false);
    }
  }

  // combine per-half partial row sums once, then normalize + store.
  lsum += __shfl_xor(lsum, 16, 32);
  float inv = 1.0f / lsum;
  float* op = attn + (hg*NTOK + (q0 + l16))*DHEAD;
#pragma unroll
  for (int r = 0; r < 8; ++r) {
    op[hp*8 + r]      = o_lo[r]*inv;
    op[16 + hp*8 + r] = o_hi[r]*inv;
  }
}

// ---------------------------------------------------------------------------
// K3: sdp1 tail (xc -> Wh1 chain) -> Wh2 -> sdp2 qkv
// ---------------------------------------------------------------------------
__global__ void __launch_bounds__(256)
k_post1(const float* __restrict__ vf, const float* __restrict__ attn,
        const float* __restrict__ W1,  const float* __restrict__ b1,
        const float* __restrict__ W10, const float* __restrict__ b10,
        const float* __restrict__ W11, const float* __restrict__ b11,
        const float* __restrict__ W2,  const float* __restrict__ b2,
        const float* __restrict__ qW,  const float* __restrict__ qb,
        _Float16* __restrict__ qh, _Float16* __restrict__ kh,
        _Float16* __restrict__ vt, float* __restrict__ vfo)
{
  __shared__ float s[10496];
  float* sW1  = s;        float* sb1  = s + 4096;
  float* sW10 = s + 4160; float* sb10 = s + 6208;
  float* sW11 = s + 6272; float* sb11 = s + 8320;
  float* sW2  = s + 8384; float* sb2  = s + 10432;
  lds_copy(sW1,  W1,  4096); lds_copy(sb1,  b1,  64);
  lds_copy(sW10, W10, 2048); lds_copy(sb10, b10, 64);
  lds_copy(sW11, W11, 2048); lds_copy(sb11, b11, 64);
  lds_copy(sW2,  W2,  2048); lds_copy(sb2,  b2,  64);
  __syncthreads();

  int t = blockIdx.x*256 + threadIdx.x;
  int b = t >> 10, n = t & 1023;

  float xc[128];
#pragma unroll
  for (int hd = 0; hd < 2; ++hd) {
    int base = ((b*2 + hd)*NTOK + n)*DHEAD;
#pragma unroll
    for (int i = 0; i < 32; ++i) {
      xc[hd*64 + i]      = vf[base + i];
      xc[hd*64 + 32 + i] = attn[base + i];
    }
  }
  float h[64], g[64];
  glinear<2,32,64,true >(xc, h, sW1,  sb1);
  glinear<2,32,32,true >(h,  g, sW10, sb10);
  glinear<2,32,32,true >(g,  h, sW11, sb11);
  glinear<2,32,32,false>(h,  g, sW2,  sb2);     // Wh2, no activation
  qkv_project_store(g, qW, qb, b, n, qh, kh, vt, vfo);
}

// ---------------------------------------------------------------------------
// K5: sdp2 tail -> x1 head (Wh3,Wh4) -> out = x0' + x1 ; stash x[:,32:] for mean
// ---------------------------------------------------------------------------
__global__ void __launch_bounds__(256)
k_post2(const float* __restrict__ x0,
        const float* __restrict__ vf, const float* __restrict__ attn,
        const float* __restrict__ W1,  const float* __restrict__ b1,
        const float* __restrict__ W10, const float* __restrict__ b10,
        const float* __restrict__ W11, const float* __restrict__ b11,
        const float* __restrict__ W3,  const float* __restrict__ b3,
        const float* __restrict__ W4,  const float* __restrict__ b4,
        float* __restrict__ out, float* __restrict__ xhi)
{
  __shared__ float s[9539];
  float* sW1  = s;        float* sb1  = s + 4096;
  float* sW10 = s + 4160; float* sb10 = s + 6208;
  float* sW11 = s + 6272; float* sb11 = s + 8320;
  float* sW3  = s + 8384; float* sb3  = s + 9408;
  float* sW4  = s + 9440; float* sb4  = s + 9536;
  lds_copy(sW1,  W1,  4096); lds_copy(sb1,  b1,  64);
  lds_copy(sW10, W10, 2048); lds_copy(sb10, b10, 64);
  lds_copy(sW11, W11, 2048); lds_copy(sb11, b11, 64);
  lds_copy(sW3,  W3,  1024); lds_copy(sb3,  b3,  32);
  lds_copy(sW4,  W4,  96);   lds_copy(sb4,  b4,  3);
  __syncthreads();

  int t = blockIdx.x*256 + threadIdx.x;
  int b = t >> 10, n = t & 1023;

  float xc[128];
#pragma unroll
  for (int hd = 0; hd < 2; ++hd) {
    int base = ((b*2 + hd)*NTOK + n)*DHEAD;
#pragma unroll
    for (int i = 0; i < 32; ++i) {
      xc[hd*64 + i]      = vf[base + i];
      xc[hd*64 + 32 + i] = attn[base + i];
    }
  }
  float h[64], g[64];
  glinear<2,32,64,true>(xc, h, sW1,  sb1);
  glinear<2,32,32,true>(h,  g, sW10, sb10);
  glinear<2,32,32,true>(g,  h, sW11, sb11);     // x

  float y1[32], y2[3];
  glinear<1,32,32,true>(h,  y1, sW3, sb3);      // uses x[0:32]
  glinear<1,3, 32,true>(y1, y2, sW4, sb4);

  out[t*3 + 0] = x0[t*3 + 0] + y2[0];
  out[t*3 + 1] = x0[t*3 + 1] + y2[1];
  out[t*3 + 2] = 1.0f/x0[t*3 + 2] + y2[2];

#pragma unroll
  for (int j = 0; j < 32; ++j) xhi[t*32 + j] = h[32 + j];
}

// ---------------------------------------------------------------------------
// K6: x2 = lrelu(Wh6 @ lrelu(Wh5 @ mean_n(x[:,32:]) + bh5) + bh6), per batch
// ---------------------------------------------------------------------------
__global__ void __launch_bounds__(256)
k_mean_head(const float* __restrict__ xhi,
            const float* __restrict__ W5, const float* __restrict__ b5,
            const float* __restrict__ W6, const float* __restrict__ b6,
            float* __restrict__ x2)
{
  __shared__ float red[256];
  __shared__ float mean[32];
  __shared__ float h5[32];
  int b = blockIdx.x;
  int c = threadIdx.x & 31, r = threadIdx.x >> 5;
  float acc = 0.0f;
  for (int n = r; n < NTOK; n += 8) acc += xhi[(b*NTOK + n)*32 + c];
  red[threadIdx.x] = acc;
  __syncthreads();
  if (r < 4) red[threadIdx.x] += red[threadIdx.x + 128];
  __syncthreads();
  if (r < 2) red[threadIdx.x] += red[threadIdx.x + 64];
  __syncthreads();
  if (r < 1) mean[c] = (red[c] + red[c + 32]) * (1.0f/NTOK);
  __syncthreads();
  if (threadIdx.x < 32) {
    float a = b5[threadIdx.x];
#pragma unroll
    for (int i = 0; i < 32; ++i) a = fmaf(W5[threadIdx.x*32 + i], mean[i], a);
    h5[threadIdx.x] = lrelu(a);
  }
  __syncthreads();
  if (threadIdx.x < 3) {
    float a = b6[threadIdx.x];
#pragma unroll
    for (int i = 0; i < 32; ++i) a = fmaf(W6[threadIdx.x*32 + i], h5[i], a);
    x2[b*3 + threadIdx.x] = lrelu(a);
  }
}

// ---------------------------------------------------------------------------
// K7: out[:,:,0:2] += x2[b,0:2]; out[:,:,2] = 1/out[:,:,2]
// ---------------------------------------------------------------------------
__global__ void __launch_bounds__(256)
k_finalize(const float* __restrict__ x2, float* __restrict__ out)
{
  int t = blockIdx.x*256 + threadIdx.x;
  if (t >= TOK) return;
  int b = t >> 10;
  out[t*3 + 0] += x2[b*3 + 0];
  out[t*3 + 1] += x2[b*3 + 1];
  out[t*3 + 2]  = 1.0f / out[t*3 + 2];
}

// ---------------------------------------------------------------------------
// Host launch. Input order assumes JAX tree-flatten of setup_inputs():
// x0 first, then params flattened with dict keys sorted (capitals before
// lowercase): Wh2..Wh6, bh2..bh6, emb.e{1..5}_{W,b}, sdp1.{Wh1,Wh1_0,Wh1_1,
// bh1,bh1_0,bh1_1,qkv_W,qkv_b}, sdp2.{same}.
// ---------------------------------------------------------------------------
extern "C" void kernel_launch(void* const* d_in, const int* in_sizes, int n_in,
                              void* d_out, int out_size, void* d_ws, size_t ws_size,
                              hipStream_t stream)
{
  (void)in_sizes; (void)n_in; (void)out_size; (void)ws_size;
  const float* x0    = (const float*)d_in[0];
  const float* Wh2   = (const float*)d_in[1];
  const float* Wh3   = (const float*)d_in[2];
  const float* Wh4   = (const float*)d_in[3];
  const float* Wh5   = (const float*)d_in[4];
  const float* Wh6   = (const float*)d_in[5];
  const float* bh2   = (const float*)d_in[6];
  const float* bh3   = (const float*)d_in[7];
  const float* bh4   = (const float*)d_in[8];
  const float* bh5   = (const float*)d_in[9];
  const float* bh6   = (const float*)d_in[10];
  const float* e1W   = (const float*)d_in[11];
  const float* e1b   = (const float*)d_in[12];
  const float* e2W   = (const float*)d_in[13];
  const float* e2b   = (const float*)d_in[14];
  const float* e3W   = (const float*)d_in[15];
  const float* e3b   = (const float*)d_in[16];
  const float* e4W   = (const float*)d_in[17];
  const float* e4b   = (const float*)d_in[18];
  const float* e5W   = (const float*)d_in[19];
  const float* e5b   = (const float*)d_in[20];
  const float* s1W1  = (const float*)d_in[21];
  const float* s1W10 = (const float*)d_in[22];
  const float* s1W11 = (const float*)d_in[23];
  const float* s1b1  = (const float*)d_in[24];
  const float* s1b10 = (const float*)d_in[25];
  const float* s1b11 = (const float*)d_in[26];
  const float* s1qW  = (const float*)d_in[27];
  const float* s1qb  = (const float*)d_in[28];
  const float* s2W1  = (const float*)d_in[29];
  const float* s2W10 = (const float*)d_in[30];
  const float* s2W11 = (const float*)d_in[31];
  const float* s2b1  = (const float*)d_in[32];
  const float* s2b10 = (const float*)d_in[33];
  const float* s2b11 = (const float*)d_in[34];
  const float* s2qW  = (const float*)d_in[35];
  const float* s2qb  = (const float*)d_in[36];

  float* out = (float*)d_out;
  char*  ws  = (char*)d_ws;

  const size_t MB = 1u << 20;
  _Float16* qh   = (_Float16*)(ws + 0*MB);    // [64][1024][32] f16 : 4 MB
  _Float16* kh   = (_Float16*)(ws + 4*MB);    // 4 MB
  _Float16* vt   = (_Float16*)(ws + 8*MB);    // [64][32][1024] f16 : 4 MB
  float*    vf1  = (float*)   (ws + 12*MB);   // [64][1024][32] f32 : 8 MB
  float*    attn = (float*)   (ws + 20*MB);   // 8 MB
  float*    vf2  = (float*)   (ws + 28*MB);   // 8 MB
  float*    xhi  = (float*)   (ws + 36*MB);   // [32768][32] f32 : 4 MB
  float*    x2b  = (float*)   (ws + 40*MB);   // [32][3]

  const int TB = TOK/256;                     // 128 token blocks
  const int AB = (NBATCH*2*(NTOK/16))/4;      // 1024 attention blocks, 4 waves each

  k_emb_qkv<<<TB, 256, 0, stream>>>(x0, e1W,e1b, e2W,e2b, e3W,e3b, e4W,e4b, e5W,e5b,
                                    s1qW, s1qb, qh, kh, vt, vf1);
  k_attn   <<<AB, 128, 0, stream>>>(qh, kh, vt, attn);
  k_post1  <<<TB, 256, 0, stream>>>(vf1, attn, s1W1,s1b1, s1W10,s1b10, s1W11,s1b11,
                                    Wh2, bh2, s2qW, s2qb, qh, kh, vt, vf2);
  k_attn   <<<AB, 128, 0, stream>>>(qh, kh, vt, attn);
  k_post2  <<<TB, 256, 0, stream>>>(x0, vf2, attn, s2W1,s2b1, s2W10,s2b10, s2W11,s2b11,
                                    Wh3, bh3, Wh4, bh4, out, xhi);
  k_mean_head<<<NBATCH, 256, 0, stream>>>(xhi, Wh5, bh5, Wh6, bh6, x2b);
  k_finalize <<<TB, 256, 0, stream>>>(x2b, out);
}
